// SAGE_15625091023094
// MI455X (gfx1250) — matrix operand order repaired and checked by reading, not compiled, not run on hardware
//
#include <hip/hip_runtime.h>
#include <math.h>

// ---------------------------------------------------------------------------
// GraphSAGE 3-layer inference for MI455X (gfx1250, wave32, WMMA).
// - Edge gather + L2-resident f32 atomics for segment-mean (HBM-bound part).
// - GEMMs: bf16 WMMA (v_wmma_f32_16x16x32_bf16), f32 accumulate.
// - Weights pre-packed once per launch to transposed bf16 [n][k]; GEMM B-tiles
//   then stream into LDS via GLOBAL_LOAD_ASYNC_TO_LDS_B128 (ASYNCcnt path).
// ---------------------------------------------------------------------------

typedef __attribute__((ext_vector_type(16))) __bf16 v16bf;
typedef __attribute__((ext_vector_type(8)))  float  v8f;

struct __align__(16) U4 { unsigned int a, b, c, d; };
struct __align__(8)  U2 { unsigned int a, b; };
union Frag { v16bf v; U4 q[2]; };

__device__ __forceinline__ unsigned short f2bf(float f) {
  union { float f; unsigned int u; } c; c.f = f;
  unsigned int r = c.u + 0x7FFFu + ((c.u >> 16) & 1u);   // round-to-nearest-even
  return (unsigned short)(r >> 16);
}

// ---------------------------------------------------------------------------
// Edge scatter: one wave32 per edge. float4 gather of the source row, f32
// atomic adds into agg (agg fits in 192MB L2, atomics resolve there) + count.
// ---------------------------------------------------------------------------
template<int DIN>
__global__ __launch_bounds__(256) void sage_scatter(
    const float* __restrict__ x, const int* __restrict__ src,
    const int* __restrict__ dst, float* __restrict__ agg,
    float* __restrict__ cnt, int E) {
  int gtid = blockIdx.x * blockDim.x + threadIdx.x;
  int edge = gtid >> 5;
  int lane = gtid & 31;
  if (edge >= E) return;
  int s = src[edge];
  int d = dst[edge];
  const float* xs = x + (size_t)s * DIN;
  float*       ag = agg + (size_t)d * DIN;
#pragma unroll
  for (int c = 0; c < DIN / 128; ++c) {
    int f = c * 128 + lane * 4;
    float4 v = *(const float4*)(xs + f);
    __hip_atomic_fetch_add(ag + f + 0, v.x, __ATOMIC_RELAXED, __HIP_MEMORY_SCOPE_AGENT);
    __hip_atomic_fetch_add(ag + f + 1, v.y, __ATOMIC_RELAXED, __HIP_MEMORY_SCOPE_AGENT);
    __hip_atomic_fetch_add(ag + f + 2, v.z, __ATOMIC_RELAXED, __HIP_MEMORY_SCOPE_AGENT);
    __hip_atomic_fetch_add(ag + f + 3, v.w, __ATOMIC_RELAXED, __HIP_MEMORY_SCOPE_AGENT);
  }
  if (lane == 0)
    __hip_atomic_fetch_add(cnt + d, 1.0f, __ATOMIC_RELAXED, __HIP_MEMORY_SCOPE_AGENT);
}

// ---------------------------------------------------------------------------
// Pre-pack [Wl; Wr] (each [Din,Dout] row-major f32) into transposed bf16
// WB[n][k], n in [0,DoutPad), k in [0,2*Din). Rows n>=Dout are zero.
// Tiny (<0.5 MB total), runs once per launch, stays L2-resident.
// ---------------------------------------------------------------------------
__global__ __launch_bounds__(256) void pack_weights(
    const float* __restrict__ Wl, const float* __restrict__ Wr,
    unsigned short* __restrict__ WB, int Din, int Dout, int DoutPad) {
  int idx = blockIdx.x * blockDim.x + threadIdx.x;
  int K2 = 2 * Din;
  if (idx >= DoutPad * K2) return;
  int n = idx / K2, k = idx - n * K2;
  float v = 0.f;
  if (n < Dout)
    v = (k < Din) ? Wl[(size_t)k * Dout + n] : Wr[(size_t)(k - Din) * Dout + n];
  WB[idx] = f2bf(v);
}

// ---------------------------------------------------------------------------
// Fused SAGE GEMM: out[m,n] = sum_k [mean | x_tgt][m,k] * [Wl; Wr][k,n] + b[n]
// K = 2*Din, mean = agg/max(cnt,1) folded into the A-tile stage.
// Block = 256 threads = 8 waves, 64x64 output tile, each wave owns 16x32
// (two 16x16x32 bf16 WMMAs per K-step of 32).
// A: f32->bf16 converted + staged via ds_store; B: bf16 rows streamed
// straight into LDS with GLOBAL_LOAD_ASYNC_TO_LDS_B128 (ASYNCcnt).
// ---------------------------------------------------------------------------
template<bool RELU>
__global__ __launch_bounds__(256) void sage_gemm(
    const float* __restrict__ agg, const float* __restrict__ cnt,
    const float* __restrict__ xin, const unsigned short* __restrict__ WB,
    const float* __restrict__ bias, float* __restrict__ out,
    int nTgt, int Din, int Dout) {
  __shared__ __align__(16) unsigned short Alds[64 * 32];
  __shared__ __align__(16) unsigned short Blds[64 * 32];

  const int tid  = threadIdx.x;
  const int wid  = tid >> 5;
  const int lane = tid & 31;
  const int lrow = lane & 15;
  const int half = lane >> 4;            // lane half selects K sub-range
  const int m0   = (wid & 3) * 16;       // 4 wave-rows
  const int n0   = (wid >> 2) * 32;      // 2 wave-cols of 32
  const int blockM = blockIdx.x * 64;
  const int blockN = blockIdx.y * 64;
  const int K2 = 2 * Din;

  v8f c0 = {};
  v8f c1 = {};

  const int numK = K2 >> 5;              // Din is a multiple of 32
  for (int kt = 0; kt < numK; ++kt) {
    const int ktBase = kt << 5;

    // ---- B tile: async copy 64 rows x 64B from packed bf16 WB[n][k] ----
    if (tid < 256) {
      int nl = tid >> 2, h = tid & 3;    // 4 x 16B per 64B row slice
      const unsigned short* srcp =
          WB + (size_t)(blockN + nl) * K2 + ktBase + h * 8;
      unsigned dstLds = (unsigned)(size_t)&Blds[nl * 32 + h * 8];
      asm volatile("global_load_async_to_lds_b128 %0, %1, off"
                   :: "v"(dstLds), "v"(srcp) : "memory");
    }

    // ---- A tile: 64 rows x 32 K, f32 -> bf16, mean folded in ----
#pragma unroll
    for (int it = 0; it < 2; ++it) {
      int e   = it * 256 + tid;
      int row = e >> 3, seg = e & 7;
      int gm  = blockM + row;
      int gk  = ktBase + seg * 4;
      float4 v = make_float4(0.f, 0.f, 0.f, 0.f);
      if (gm < nTgt) {
        if (gk < Din) {                  // mean part (agg / max(cnt,1))
          float ic = 1.0f / fmaxf(cnt[gm], 1.0f);
          float4 t = *(const float4*)(agg + (size_t)gm * Din + gk);
          v.x = t.x * ic; v.y = t.y * ic; v.z = t.z * ic; v.w = t.w * ic;
          if (gk + 32 < Din)
            __builtin_prefetch(agg + (size_t)gm * Din + gk + 32, 0, 1);
        } else {                         // root (x[:n_tgt]) part
          int gx = gk - Din;
          v = *(const float4*)(xin + (size_t)gm * Din + gx);
          if (gx + 32 < Din)
            __builtin_prefetch(xin + (size_t)gm * Din + gx + 32, 0, 1);
        }
      }
      U2 p;
      p.a = (unsigned)f2bf(v.x) | ((unsigned)f2bf(v.y) << 16);
      p.b = (unsigned)f2bf(v.z) | ((unsigned)f2bf(v.w) << 16);
      *(U2*)&Alds[row * 32 + seg * 4] = p;
    }

    // async B copies must have landed in LDS before the barrier
    asm volatile("s_wait_asynccnt 0x0" ::: "memory");
    __syncthreads();

    // ---- fragments (ISA 16-bit A/B layouts: lane halves split K) ----
    Frag a, b0, b1;
    const unsigned short* arow = &Alds[(m0 + lrow) * 32 + half * 8];
    a.q[0] = *(const U4*)(arow);         // lanes<16: K0-7   | lanes>=16: K8-15
    a.q[1] = *(const U4*)(arow + 16);    // lanes<16: K16-23 | lanes>=16: K24-31
    const unsigned short* brow0 = &Blds[(n0 + lrow) * 32 + half * 8];
    b0.q[0] = *(const U4*)(brow0);       // same K split as A
    b0.q[1] = *(const U4*)(brow0 + 16);
    const unsigned short* brow1 = brow0 + 16 * 32;
    b1.q[0] = *(const U4*)(brow1);
    b1.q[1] = *(const U4*)(brow1 + 16);

    c0 = __builtin_amdgcn_wmma_f32_16x16x32_bf16(false, a.v, false, b0.v,
                                                 (short)0, c0, false, false);
    c1 = __builtin_amdgcn_wmma_f32_16x16x32_bf16(false, a.v, false, b1.v,
                                                 (short)0, c1, false, false);
    __syncthreads();
  }

  // ---- epilogue: +bias, optional relu; C/D layout VGPR r -> M=r / M=8+r ----
  const int mBase = blockM + m0 + half * 8;
  const int nBase = blockN + n0 + lrow;
#pragma unroll
  for (int r = 0; r < 8; ++r) {
    int M = mBase + r;
    if (M < nTgt) {
      int N = nBase;
      if (N < Dout) {
        float v = c0[r] + bias[N];
        if (RELU) v = fmaxf(v, 0.f);
        out[(size_t)M * Dout + N] = v;
      }
      N = nBase + 16;
      if (N < Dout) {
        float v = c1[r] + bias[N];
        if (RELU) v = fmaxf(v, 0.f);
        out[(size_t)M * Dout + N] = v;
      }
    }
  }
}

// ---------------------------------------------------------------------------
// log_softmax: one wave32 per row, shfl-xor reductions.
// ---------------------------------------------------------------------------
__global__ __launch_bounds__(256) void lsm_kernel(
    const float* __restrict__ in, float* __restrict__ out, int rows, int C) {
  int wid  = blockIdx.x * (blockDim.x >> 5) + (threadIdx.x >> 5);
  int lane = threadIdx.x & 31;
  if (wid >= rows) return;
  const float* r = in + (size_t)wid * C;
  float m = -INFINITY;
  for (int j = lane; j < C; j += 32) m = fmaxf(m, r[j]);
#pragma unroll
  for (int o = 16; o > 0; o >>= 1) m = fmaxf(m, __shfl_xor(m, o, 32));
  float s = 0.f;
  for (int j = lane; j < C; j += 32) s += expf(r[j] - m);
#pragma unroll
  for (int o = 16; o > 0; o >>= 1) s += __shfl_xor(s, o, 32);
  float ls = logf(s);
  for (int j = lane; j < C; j += 32)
    out[(size_t)wid * C + j] = (r[j] - m) - ls;
}

// ---------------------------------------------------------------------------
// Launch. Inputs (in order): x, src0, dst0, src1, dst1, src2, dst2,
// Wl0, Wr0, b0, Wl1, Wr1, b1, Wl2, Wr2, b2, size1, size2, size3.
// ---------------------------------------------------------------------------
extern "C" void kernel_launch(void* const* d_in, const int* in_sizes, int n_in,
                              void* d_out, int out_size, void* d_ws, size_t ws_size,
                              hipStream_t stream) {
  const float* x    = (const float*)d_in[0];
  const int*   src0 = (const int*)d_in[1];
  const int*   dst0 = (const int*)d_in[2];
  const int*   src1 = (const int*)d_in[3];
  const int*   dst1 = (const int*)d_in[4];
  const int*   src2 = (const int*)d_in[5];
  const int*   dst2 = (const int*)d_in[6];
  const float* Wl0  = (const float*)d_in[7];
  const float* Wr0  = (const float*)d_in[8];
  const float* b0   = (const float*)d_in[9];
  const float* Wl1  = (const float*)d_in[10];
  const float* Wr1  = (const float*)d_in[11];
  const float* b1   = (const float*)d_in[12];
  const float* Wl2  = (const float*)d_in[13];
  const float* Wr2  = (const float*)d_in[14];
  const float* b2   = (const float*)d_in[15];

  const int E0 = in_sizes[1], E1 = in_sizes[3], E2 = in_sizes[5];
  const int N1 = 100000, N2 = 10000, N3 = 1000;     // size1/2/3 (fixed dispatch)
  const int DIN = 128, DH = 256;
  const int DOUT = in_sizes[15];                    // 47
  const int DOUTP = ((DOUT + 63) / 64) * 64;        // 64

  // workspace layout (floats); zeroed region first
  float* ws = (float*)d_ws;
  size_t o = 0;
  float* agg0 = ws + o; o += (size_t)N1 * DIN;
  float* cnt0 = ws + o; o += N1;
  float* agg1 = ws + o; o += (size_t)N2 * DH;
  float* cnt1 = ws + o; o += N2;
  float* agg2 = ws + o; o += (size_t)N3 * DH;
  float* cnt2 = ws + o; o += N3;
  size_t zeroFloats = o;
  float* h1 = ws + o; o += (size_t)N1 * DH;
  float* h2 = ws + o; o += (size_t)N2 * DH;
  float* t3 = ws + o; o += (size_t)N3 * DOUT;
  o = (o + 3) & ~(size_t)3;                         // 16B align for bf16 packs
  unsigned short* WB0 = (unsigned short*)(ws + o);  // [DH][2*DIN]
  unsigned short* WB1 = WB0 + (size_t)DH * (2 * DIN);    // [DH][2*DH]
  unsigned short* WB2 = WB1 + (size_t)DH * (2 * DH);     // [DOUTP][2*DH]

  hipMemsetAsync(d_ws, 0, zeroFloats * sizeof(float), stream);

  // pack all weight matrices to transposed bf16 (independent of graph data)
  pack_weights<<<(DH * 2 * DIN + 255) / 256, 256, 0, stream>>>(Wl0, Wr0, WB0, DIN, DH, DH);
  pack_weights<<<(DH * 2 * DH + 255) / 256, 256, 0, stream>>>(Wl1, Wr1, WB1, DH, DH, DH);
  pack_weights<<<(DOUTP * 2 * DH + 255) / 256, 256, 0, stream>>>(Wl2, Wr2, WB2, DH, DOUT, DOUTP);

  // layer 0: x[600000,128] -> h1[100000,256], relu
  sage_scatter<128><<<(E0 + 7) / 8, 256, 0, stream>>>(x, src0, dst0, agg0, cnt0, E0);
  sage_gemm<true><<<dim3((N1 + 63) / 64, DH / 64), 256, 0, stream>>>(
      agg0, cnt0, x, WB0, b0, h1, N1, DIN, DH);

  // layer 1: h1 -> h2[10000,256], relu
  sage_scatter<256><<<(E1 + 7) / 8, 256, 0, stream>>>(h1, src1, dst1, agg1, cnt1, E1);
  sage_gemm<true><<<dim3((N2 + 63) / 64, DH / 64), 256, 0, stream>>>(
      agg1, cnt1, h1, WB1, b1, h2, N2, DH, DH);

  // layer 2: h2 -> t3[1000,47], no relu
  sage_scatter<256><<<(E2 + 7) / 8, 256, 0, stream>>>(h2, src2, dst2, agg2, cnt2, E2);
  sage_gemm<false><<<dim3((N3 + 63) / 64, DOUTP / 64), 256, 0, stream>>>(
      agg2, cnt2, h2, WB2, b2, t3, N3, DH, DOUT);

  // log_softmax -> d_out
  lsm_kernel<<<(N3 + 7) / 8, 256, 0, stream>>>(t3, (float*)d_out, N3, DOUT);
}
